// SAGE_66297115181594
// MI455X (gfx1250) — matrix operand order repaired and checked by reading, not compiled, not run on hardware
//
#include <hip/hip_runtime.h>

// ---------------------------------------------------------------------------
// GraphSAGE 2-layer forward for MI455X (gfx1250, wave32, WMMA).
//   layer: h = relu( (segsum(x[src])/deg) @ W_l + b_l + x @ W_r )
// Aggregation is L2-resident scatter/gather with f32 atomics; dense GEMMs use
// v_wmma_f32_16x16x32_bf16 with fragment-packed bf16 weights.
// ---------------------------------------------------------------------------

#define N_NODES  50000
#define N_EDGES  800000
#define DIM      128
#define NUM_ROOT 10000

typedef __attribute__((ext_vector_type(16))) __bf16 v16bf;
typedef __attribute__((ext_vector_type(8)))  float  v8f;

// ---------------------------------------------------------------------------
__global__ void zero_f32(float* __restrict__ p, long n) {
    long t = (long)blockIdx.x * blockDim.x + threadIdx.x;
    if (t < n) p[t] = 0.0f;
}

// deg[dst] += 1  (float counts are exact far beyond max degree here)
__global__ void edge_deg(const int* __restrict__ dst, float* __restrict__ deg) {
    int e = blockIdx.x * blockDim.x + threadIdx.x;
    if (e < N_EDGES) atomicAdd(deg + dst[e], 1.0f);
}

// in-place deg -> inv_deg  (reference: deg>0 ? 1/deg : 0)
__global__ void make_inv(float* __restrict__ deg) {
    int i = blockIdx.x * blockDim.x + threadIdx.x;
    if (i < N_NODES) {
        float d = deg[i];
        deg[i] = d > 0.0f ? 1.0f / d : 0.0f;
    }
}

// agg[dst] += feat[src]; 32 lanes per edge, float4 per lane (coalesced 512B
// row reads, contiguous atomic bursts). max_dst prunes layer-1 work to roots.
__global__ void scatter_edges(const float* __restrict__ feat,
                              const int* __restrict__ src,
                              const int* __restrict__ dst,
                              float* __restrict__ agg, int max_dst) {
    long t = (long)blockIdx.x * blockDim.x + threadIdx.x;
    if (t >= (long)N_EDGES * 32) return;
    int e = (int)(t >> 5);
    int g = ((int)t & 31) << 2;            // dim group of 4
    int d = dst[e];
    if (d >= max_dst) return;
    int s = src[e];
    const float4 v = *(const float4*)(feat + (size_t)s * DIM + g);
    float* p = agg + (size_t)d * DIM + g;
    atomicAdd(p + 0, v.x);
    atomicAdd(p + 1, v.y);
    atomicAdd(p + 2, v.z);
    atomicAdd(p + 3, v.w);
}

// Repack four 128x128 f32 weights into WMMA B-fragment order (bf16).
// B layout (ISA 7.12.2, 16-bit B 32x16): lane = N (lane&15), half-wave picks
// K+8; element j of fragment (nt,kc,lane) = W[kc*32 + (j/8)*16 + hi*8 + j%8][nt*16 + n].
__global__ void pack_weights(const float* __restrict__ Wl0,
                             const float* __restrict__ Wr0,
                             const float* __restrict__ Wl1,
                             const float* __restrict__ Wr1,
                             __bf16* __restrict__ bp) {
    int t = blockIdx.x * blockDim.x + threadIdx.x;
    if (t >= 4 * 8 * 4 * 32 * 16) return;
    const float* W;
    switch (t >> 14) {
        case 0:  W = Wl0; break;
        case 1:  W = Wr0; break;
        case 2:  W = Wl1; break;
        default: W = Wr1; break;
    }
    int u    = t & 16383;
    int j    = u & 15;
    int lane = (u >> 4) & 31;
    int kc   = (u >> 9) & 3;
    int nt   = u >> 11;
    int kb   = kc * 32 + ((j >> 3) << 4) + ((lane >> 4) << 3) + (j & 7);   // K index
    int n    = (nt << 4) + (lane & 15);            // N index
    bp[t] = (__bf16)W[kb * DIM + n];
}

// One block = 16-row M-tile; 8 waves, one 16-col N-tile each.
// acc = (agg*inv_deg) @ W_l + x @ W_r  via chained bf16 WMMAs, then +bias/ReLU.
__global__ __launch_bounds__(256) void sage_gemm(
        const float* __restrict__ aggm, const float* __restrict__ invd,
        const float* __restrict__ xr,
        const __bf16* __restrict__ bpL, const __bf16* __restrict__ bpR,
        const float* __restrict__ bias, float* __restrict__ out, int do_relu) {
    const int lane = threadIdx.x & 31;
    const int nt   = threadIdx.x >> 5;     // N-tile owned by this wave
    const int m    = lane & 15;
    const int hi   = lane >> 4;            // half-wave selector
    const int row0 = blockIdx.x << 4;
    const int row  = row0 + m;

    const float s = invd[row];
    const float* __restrict__ ag = aggm + (size_t)row * DIM;
    const float* __restrict__ xv = xr   + (size_t)row * DIM;

    v8f acc = {};
#pragma unroll
    for (int kc = 0; kc < 4; ++kc) {
        const int kb = kc * 32 + hi * 8;   // A fragment K base for this lane
        v16bf aL, aR;
#pragma unroll
        for (int j = 0; j < 8; ++j) {
            aL[j]     = (__bf16)(ag[kb + j]      * s);
            aL[j + 8] = (__bf16)(ag[kb + 16 + j] * s);
            aR[j]     = (__bf16)(xv[kb + j]);
            aR[j + 8] = (__bf16)(xv[kb + 16 + j]);
        }
        const v16bf bL = *(const v16bf*)(bpL + (((nt * 4 + kc) * 32 + lane) << 4));
        const v16bf bR = *(const v16bf*)(bpR + (((nt * 4 + kc) * 32 + lane) << 4));
        acc = __builtin_amdgcn_wmma_f32_16x16x32_bf16(false, aL, false, bL,
                                                      (short)0, acc, false, false);
        acc = __builtin_amdgcn_wmma_f32_16x16x32_bf16(false, aR, false, bR,
                                                      (short)0, acc, false, false);
    }

    const int   n  = (nt << 4) + m;        // output column for this lane
    const float bv = bias[n];
#pragma unroll
    for (int r = 0; r < 8; ++r) {          // C/D layout: VGPR r -> row hi*8+r
        float v = acc[r] + bv;
        if (do_relu) v = fmaxf(v, 0.0f);
        out[(size_t)(row0 + hi * 8 + r) * DIM + n] = v;
    }
}

// ---------------------------------------------------------------------------
extern "C" void kernel_launch(void* const* d_in, const int* in_sizes, int n_in,
                              void* d_out, int out_size, void* d_ws, size_t ws_size,
                              hipStream_t stream) {
    const float* x   = (const float*)d_in[0];
    const int*   nbr = (const int*)  d_in[1];
    const float* Wl0 = (const float*)d_in[2];
    const float* bl0 = (const float*)d_in[3];
    const float* Wr0 = (const float*)d_in[4];
    const float* Wl1 = (const float*)d_in[5];
    const float* bl1 = (const float*)d_in[6];
    const float* Wr1 = (const float*)d_in[7];
    const int* src = nbr;
    const int* dst = nbr + N_EDGES;
    float* out = (float*)d_out;

    // Workspace layout (floats): deg/inv_deg | agg | h | packed bf16 weights
    float*  ws    = (float*)d_ws;
    float*  deg   = ws;                                   // N_NODES
    float*  agg   = ws + 50048;                           // N_NODES*DIM
    float*  h     = agg + (size_t)N_NODES * DIM;          // N_NODES*DIM
    __bf16* bpack = (__bf16*)(h + (size_t)N_NODES * DIM); // 4 * 16384 bf16
    __bf16* bWl0 = bpack;
    __bf16* bWr0 = bpack + 16384;
    __bf16* bWl1 = bpack + 2 * 16384;
    __bf16* bWr1 = bpack + 3 * 16384;

    const long nz0 = 50048 + (long)N_NODES * DIM;         // deg + agg contiguous
    zero_f32<<<(int)((nz0 + 255) / 256), 256, 0, stream>>>(deg, nz0);
    pack_weights<<<(4 * 16384 + 255) / 256, 256, 0, stream>>>(Wl0, Wr0, Wl1, Wr1, bpack);

    edge_deg<<<(N_EDGES + 255) / 256, 256, 0, stream>>>(dst, deg);
    make_inv<<<(N_NODES + 255) / 256, 256, 0, stream>>>(deg);

    const long nsc = (long)N_EDGES * 32;
    // Layer 0
    scatter_edges<<<(int)((nsc + 255) / 256), 256, 0, stream>>>(x, src, dst, agg, N_NODES);
    sage_gemm<<<N_NODES / 16, 256, 0, stream>>>(agg, deg, x, bWl0, bWr0, bl0, h, 1);

    // Layer 1 (only root rows ultimately needed)
    const long nz1 = (long)N_NODES * DIM;
    zero_f32<<<(int)((nz1 + 255) / 256), 256, 0, stream>>>(agg, nz1);
    scatter_edges<<<(int)((nsc + 255) / 256), 256, 0, stream>>>(h, src, dst, agg, NUM_ROOT);
    sage_gemm<<<NUM_ROOT / 16, 256, 0, stream>>>(agg, deg, h, bWl1, bWr1, bl1, out, 0);
}
